// GNNLayer_20547123544556
// MI455X (gfx1250) — matrix operand order repaired and checked by reading, not compiled, not run on hardware
//
#include <hip/hip_runtime.h>
#include <stdint.h>

// -----------------------------------------------------------------------------
// GNN layer: h2 = X @ (W A)^T + b, A = I + S + S^T (8-neighbor stencil with
// duplicated interior edges -> coeff 2). A folded analytically into W (prep_w).
// X and W*A split into bf16 hi+lo planes; GEMM uses v_wmma_f32_16x16x32_bf16
// (3 WMMAs per chunk: hi*hi + hi*lo + lo*hi ~ f32 accuracy).
//
// GEMM: 16-wave workgroup computes a 64x64 tile region; A/B chunk fragments are
// staged into LDS once per workgroup via global_load_async_to_lds_b128
// (ASYNCcnt + s_wait_asynccnt, double buffered) instead of 16 private L2 reads.
// Split-K over 32 segments + deterministic reduction (no float atomics).
// -----------------------------------------------------------------------------

#define LATN   256
#define NN     65536        // LONG*LAT
#define BB     256          // batch
#define OUTN   256          // 16*16 outputs
#define KSEG   32
#define KLEN   (NN / KSEG)  // 2048
#define NCHUNK (KLEN / 32)  // 64 K-chunks of 32 per segment

// LDS staging: 4 planes (Ahi, Alo, Bhi, Blo), 64 rows x 32 bf16, row stride
// padded 64B->80B for conflict-free ds_load_b128; double buffered.
#define ROWB   80
#define PLANEB (64 * ROWB)      // 5120 B
#define BUFB   (4 * PLANEB)     // 20480 B
#define LDSB   (2 * BUFB)       // 40960 B

typedef __bf16 bf16_t;
typedef __attribute__((ext_vector_type(16))) bf16_t v16bf;
typedef __attribute__((ext_vector_type(8)))  bf16_t v8bf;
typedef __attribute__((ext_vector_type(8)))  float  v8f;

__device__ __forceinline__ unsigned short f32_to_bf16_rne(float f) {
    union { float f; unsigned int u; } cv; cv.f = f;
    unsigned int u = cv.u;
    unsigned int r = u + 0x7FFFu + ((u >> 16) & 1u);   // round-to-nearest-even
    return (unsigned short)(r >> 16);
}
__device__ __forceinline__ float bf16_to_f32(unsigned short h) {
    union { unsigned int u; float f; } cv; cv.u = ((unsigned int)h) << 16;
    return cv.f;
}

// ---- split X into bf16 hi/lo planes ----------------------------------------
__global__ void prep_x(const float* __restrict__ x,
                       unsigned short* __restrict__ xhi,
                       unsigned short* __restrict__ xlo) {
    size_t i = (size_t)blockIdx.x * blockDim.x + threadIdx.x;   // < BB*NN
    float v = x[i];
    unsigned short hi = f32_to_bf16_rne(v);
    float res = v - bf16_to_f32(hi);
    xhi[i] = hi;
    xlo[i] = f32_to_bf16_rne(res);
}

// ---- fold A into W (9-point stencil, coeff = interior(n)+interior(n+off)),
//      then split into bf16 hi/lo --------------------------------------------
__global__ void prep_w(const float* __restrict__ W,
                       unsigned short* __restrict__ whi,
                       unsigned short* __restrict__ wlo) {
    size_t i = (size_t)blockIdx.x * blockDim.x + threadIdx.x;   // < OUTN*NN
    int n = (int)(i & (size_t)(NN - 1));
    size_t rowbase = i - (size_t)n;
    float acc = W[i];
    const int offs[8] = {-1, 1, LATN, -LATN, LATN - 1, LATN + 1, -LATN - 1, -LATN + 1};
    const int lo = LATN + 1;          // 257
    const int hi = NN - LATN - 2;     // 65278
    bool cin = (n >= lo) && (n <= hi);
#pragma unroll
    for (int j = 0; j < 8; ++j) {
        int m2 = n + offs[j];
        int coef = (cin ? 1 : 0) + ((m2 >= lo && m2 <= hi) ? 1 : 0);
        if (coef) acc += (float)coef * W[rowbase + (size_t)m2];
    }
    unsigned short h = f32_to_bf16_rne(acc);
    float res = acc - bf16_to_f32(h);
    whi[i] = h;
    wlo[i] = f32_to_bf16_rne(res);
}

// ---- main GEMM --------------------------------------------------------------
// Workgroup: 512 threads = 16 waves as a 4x4 tile grid over a 64x64 output
// region. Per K-chunk, the whole workgroup async-copies Ahi/Alo/Bhi/Blo
// (64x32 bf16 each) into LDS (2 global_load_async_to_lds_b128 per wave),
// double buffered; each wave then ds_loads its WMMA fragments.
// Fragment layouts follow ISA 7.12.2 (16-bit A 16x32 / B 32x16 / f32 C 16x16).
__global__ __launch_bounds__(512) void gemm_tile(
    const unsigned short* __restrict__ Xhi,
    const unsigned short* __restrict__ Xlo,
    const unsigned short* __restrict__ Whi,
    const unsigned short* __restrict__ Wlo,
    float* __restrict__ part) {
    __shared__ char sm[LDSB];

    const int tid  = threadIdx.x;
    const int lane = tid & 31;
    const int wave = tid >> 5;
    const int wm   = wave >> 2;        // tile row within workgroup (0..3)
    const int wn   = wave & 3;         // tile col within workgroup (0..3)
    const int l16  = lane & 15;
    const int half = (lane >> 4) & 1;
    const int bm = blockIdx.x, bn = blockIdx.y, ks = blockIdx.z;
    const int k0 = ks * KLEN;

    // --- per-thread async copy assignment: 2 x b128 ops per chunk ---
    // op o in {tid, tid+512}: plane p = o>>8, row r = (o>>2)&63, 16B chunk c = o&3
    const unsigned short* gsrc[2];
    unsigned loff[2];
#pragma unroll
    for (int q = 0; q < 2; ++q) {
        int o = tid + q * 512;
        int p = o >> 8, r = (o >> 2) & 63, c = o & 3;
        const unsigned short* plane = (p == 0) ? Xhi : (p == 1) ? Xlo
                                     : (p == 2) ? Whi : Wlo;
        int row = ((p < 2) ? bm : bn) * 64 + r;
        gsrc[q] = plane + (size_t)row * NN + (size_t)(k0 + c * 8);
        loff[q] = (unsigned)(p * PLANEB + r * ROWB + c * 16);
    }
    const unsigned ldsbase = (unsigned)(uintptr_t)(void*)sm;  // LDS offset = addr[31:0]

    auto issue = [&](int buf, int chunk) {
        unsigned lb = ldsbase + (unsigned)(buf * BUFB);
#pragma unroll
        for (int q = 0; q < 2; ++q) {
            unsigned laddr = lb + loff[q];
            uint64_t ga = (uint64_t)(uintptr_t)(gsrc[q] + (size_t)chunk * 32);
            asm volatile("global_load_async_to_lds_b128 %0, %1, off"
                         :: "v"(laddr), "v"(ga) : "memory");
        }
    };

    v8f c = {};

    issue(0, 0);
    for (int it = 0; it < NCHUNK; ++it) {
        const int buf = it & 1;
        asm volatile("s_wait_asynccnt 0x0" ::: "memory");  // my copies landed
        __syncthreads();                                   // everyone's landed
        if (it + 1 < NCHUNK) issue(buf ^ 1, it + 1);

        const char* bb = sm + buf * BUFB;
        const int aoff = (wm * 16 + l16) * ROWB + half * 16;
        const int boff = (wn * 16 + l16) * ROWB + half * 16;

        union { v16bf v; v8bf h[2]; } ahi, alo, bhi, blo;
        ahi.h[0] = *(const v8bf*)(bb + 0 * PLANEB + aoff);
        ahi.h[1] = *(const v8bf*)(bb + 0 * PLANEB + aoff + 32);
        alo.h[0] = *(const v8bf*)(bb + 1 * PLANEB + aoff);
        alo.h[1] = *(const v8bf*)(bb + 1 * PLANEB + aoff + 32);
        bhi.h[0] = *(const v8bf*)(bb + 2 * PLANEB + boff);
        bhi.h[1] = *(const v8bf*)(bb + 2 * PLANEB + boff + 32);
        blo.h[0] = *(const v8bf*)(bb + 3 * PLANEB + boff);
        blo.h[1] = *(const v8bf*)(bb + 3 * PLANEB + boff + 32);

        c = __builtin_amdgcn_wmma_f32_16x16x32_bf16(false, ahi.v, false, bhi.v,
                                                    (short)0, c, false, false);
        c = __builtin_amdgcn_wmma_f32_16x16x32_bf16(false, ahi.v, false, blo.v,
                                                    (short)0, c, false, false);
        c = __builtin_amdgcn_wmma_f32_16x16x32_bf16(false, alo.v, false, bhi.v,
                                                    (short)0, c, false, false);
        __syncthreads();   // all waves done reading buf before it is refilled
    }

    float* dst = part + (size_t)ks * (BB * OUTN)
               + (size_t)(bm * 64 + wm * 16 + half * 8) * OUTN
               + (size_t)(bn * 64 + wn * 16 + l16);
#pragma unroll
    for (int g = 0; g < 8; ++g) dst[(size_t)g * OUTN] = c[g];
}

// ---- deterministic split-K reduction + bias ---------------------------------
__global__ void reduce_out(const float* __restrict__ part,
                           const float* __restrict__ bias,
                           float* __restrict__ out) {
    int i = blockIdx.x * blockDim.x + threadIdx.x;   // < BB*OUTN
    float acc = bias[i & (OUTN - 1)];
#pragma unroll
    for (int s = 0; s < KSEG; ++s) acc += part[(size_t)s * (BB * OUTN) + i];
    out[i] = acc;
}

extern "C" void kernel_launch(void* const* d_in, const int* in_sizes, int n_in,
                              void* d_out, int out_size, void* d_ws, size_t ws_size,
                              hipStream_t stream) {
    const float* x = (const float*)d_in[0];   // (256, 256, 256) f32
    const float* W = (const float*)d_in[1];   // (256, 65536)    f32
    const float* b = (const float*)d_in[2];   // (256,)          f32
    // d_in[3]=rows, d_in[4]=cols: stencil structure derived analytically.
    float* out = (float*)d_out;

    char* ws = (char*)d_ws;
    unsigned short* Xhi = (unsigned short*)(ws);                        // 32 MB
    unsigned short* Xlo = (unsigned short*)(ws + ((size_t)32 << 20));   // 32 MB
    unsigned short* Whi = (unsigned short*)(ws + ((size_t)64 << 20));   // 32 MB
    unsigned short* Wlo = (unsigned short*)(ws + ((size_t)96 << 20));   // 32 MB
    float*          prt = (float*)(ws + ((size_t)128 << 20));           //  8 MB

    prep_x<<<(BB * NN) / 256, 256, 0, stream>>>(x, Xhi, Xlo);
    prep_w<<<(OUTN * NN) / 256, 256, 0, stream>>>(W, Whi, Wlo);
    gemm_tile<<<dim3(4, 4, KSEG), 512, 0, stream>>>(Xhi, Xlo, Whi, Wlo, prt);
    reduce_out<<<(BB * OUTN) / 256, 256, 0, stream>>>(prt, b, out);
}
